// GSEvolution_4561255268515
// MI455X (gfx1250) — matrix operand order repaired and checked by reading, not compiled, not run on hardware
//
#include <hip/hip_runtime.h>
#include <hip/hip_bf16.h>
#include <stdint.h>

// ---------------------------------------------------------------------------
// Problem constants (from reference): S=128, T=128, V=2048, OT=32, BS=32
// ---------------------------------------------------------------------------
#define SS   128
#define TT   128
#define VV   2048
#define OT   32
#define KTOT 16384   // S*T

typedef __attribute__((ext_vector_type(16))) __bf16 v16bf;
typedef __attribute__((ext_vector_type(8)))  float  v8f;

union Frag { uint4 u[2]; v16bf v; };

__device__ __forceinline__ unsigned short f2bf(float f) {
    unsigned u = __float_as_uint(f);
    u += 0x7FFFu + ((u >> 16) & 1u);          // round-to-nearest-even
    return (unsigned short)(u >> 16);
}
__device__ __forceinline__ float bf2f(unsigned short h) {
    return __uint_as_float(((unsigned)h) << 16);
}
// two bf16 in a dword -> two bf16 {1.0 if >0 else 0.0}; branch-free SWAR.
__device__ __forceinline__ unsigned mask2(unsigned w) {
    unsigned nz  = ((w & 0x7FFF7FFFu) + 0x7FFF7FFFu) & 0x80008000u; // |x| != 0
    unsigned pos = ~w & 0x80008000u;                                // sign clear
    return ((nz & pos) >> 15) * 0x3F80u;                            // 1.0bf16 per half
}

// async global -> LDS copy of 32 bytes per lane (2 x b128), ASYNCcnt tracked.
// the offset: field is added to both the LDS and the global address
__device__ __forceinline__ void async_fill32(unsigned lds, const void* g) {
    asm volatile("global_load_async_to_lds_b128 %0, %1, off"
                 :: "v"(lds), "v"(g) : "memory");
    asm volatile("global_load_async_to_lds_b128 %0, %1, off offset:16"
                 :: "v"(lds), "v"(g) : "memory");
}
__device__ __forceinline__ void wait_async0() {
    asm volatile("s_wait_asynccnt 0x0" ::: "memory");
}
// generic pointer to shared memory -> raw LDS byte offset (addr[31:0])
__device__ __forceinline__ unsigned lds_off(const void* p) {
    return (unsigned)(size_t)p;
}

// ---------------------------------------------------------------------------
// Generic NT bf16 GEMM:  C[m,n] = sum_k Arows[m][k] * Brows[n][k]
// block tile 128x128, 8 waves (2x4), wave tile 64x32, K-step 32,
// double-buffered LDS, tensor-async fills, global_prefetch, 1 barrier/iter.
// MODE 0: B is masked (x>0 -> 1.0), store fp32 to outF         (adjacency)
// MODE 1: scale by 1/(aux[m]*aux[n]), store bf16               (cosine sim)
// MODE 2: add aux[m*N+n], store bf16                           (+cosine_bias)
// MODE 3: add aux[n], store bf16                               (+adj_bias, transposed out)
// MODE 4: add aux[m&31], store fp32 to outF                    (+time_bias -> d_out)
// ---------------------------------------------------------------------------
#define LDSP 40   // padded row stride in ushorts (80B: 16B aligned, conflict-free)

template <int MODE, int LD, int K>
__global__ __launch_bounds__(256)
void gemm_nt(const unsigned short* __restrict__ Arows,
             const unsigned short* __restrict__ Brows,
             float* __restrict__ outF, unsigned short* __restrict__ outB,
             const float* __restrict__ aux)
{
    __shared__ unsigned short lA[2][128 * LDSP];
    __shared__ unsigned short lB[2][128 * LDSP];

    const int tid  = threadIdx.x;
    const int lane = tid & 31;
    const int wave = tid >> 5;        // 0..7
    const int wm   = wave >> 2;       // 0..1  (M half)
    const int wn   = wave & 3;        // 0..3  (N quarter)
    const int h    = lane >> 4;       // lane half
    const int m16  = lane & 15;

    const int rowbase = blockIdx.y * 128;
    const int colbase = blockIdx.x * 128;

    v8f acc[4][2];
#pragma unroll
    for (int i = 0; i < 4; ++i)
#pragma unroll
        for (int j = 0; j < 2; ++j)
#pragma unroll
            for (int e = 0; e < 8; ++e) acc[i][j][e] = 0.0f;

    const int sr = tid >> 1;            // staging row 0..127
    const int sp = (tid & 1) * 16;      // staging k-offset (0 or 16)

    const unsigned short* gA = Arows + (size_t)(rowbase + sr) * LD + sp;
    const unsigned short* gB = Brows + (size_t)(colbase + sr) * LD + sp;

    const unsigned ldsA0 = lds_off(&lA[0][sr * LDSP + sp]);
    const unsigned ldsA1 = lds_off(&lA[1][sr * LDSP + sp]);
    const unsigned ldsB0 = lds_off(&lB[0][sr * LDSP + sp]);
    const unsigned ldsB1 = lds_off(&lB[1][sr * LDSP + sp]);

    uint4 rb0, rb1;                     // register stage for masked B (MODE 0)
    auto loadB = [&](int k0) {
        const uint4* b = (const uint4*)(gB + k0);
        rb0 = b[0]; rb1 = b[1];
    };
    auto maskStoreB = [&](int buf) {
        uint4 b0 = rb0, b1 = rb1;
        b0.x = mask2(b0.x); b0.y = mask2(b0.y); b0.z = mask2(b0.z); b0.w = mask2(b0.w);
        b1.x = mask2(b1.x); b1.y = mask2(b1.y); b1.z = mask2(b1.z); b1.w = mask2(b1.w);
        uint4* t = (uint4*)&lB[buf][sr * LDSP + sp];
        t[0] = b0; t[1] = b1;
    };

    // ---- prologue: fill buffer 0 ----
    async_fill32(ldsA0, gA);
    if (MODE == 0) { loadB(0); maskStoreB(0); }
    else           async_fill32(ldsB0, gB);
    wait_async0();
    __syncthreads();

    int buf = 0;
    for (int k0 = 0; k0 < K; k0 += 32) {
        const bool more = (k0 + 32 < K);
        if (more) {
            async_fill32(buf ? ldsA0 : ldsA1, gA + k0 + 32);
            if (MODE == 0) loadB(k0 + 32);
            else           async_fill32(buf ? ldsB0 : ldsB1, gB + k0 + 32);
            if (k0 + 64 < K) {                       // cover HBM latency
                __builtin_prefetch(gA + k0 + 64, 0, 1);
                __builtin_prefetch(gB + k0 + 64, 0, 1);
            }
        }

        // ---- A fragments: lane holds row m16, K = (e>>3)*16 + h*8 + (e&7) ----
        Frag fa[4], fb[2];
#pragma unroll
        for (int mi = 0; mi < 4; ++mi) {
            const unsigned short* p = &lA[buf][(wm * 64 + mi * 16 + m16) * LDSP + h * 8];
            fa[mi].u[0] = *(const uint4*)p;          // K h*8 .. h*8+7
            fa[mi].u[1] = *(const uint4*)(p + 16);   // K 16+h*8 .. 16+h*8+7
        }
        // ---- B fragments: lane holds column m16, K = (e>>3)*16 + h*8 + (e&7) ----
#pragma unroll
        for (int ni = 0; ni < 2; ++ni) {
            const unsigned short* p = &lB[buf][(wn * 32 + ni * 16 + m16) * LDSP + h * 8];
            fb[ni].u[0] = *(const uint4*)p;
            fb[ni].u[1] = *(const uint4*)(p + 16);
        }
#pragma unroll
        for (int mi = 0; mi < 4; ++mi)
#pragma unroll
            for (int ni = 0; ni < 2; ++ni)
                acc[mi][ni] = __builtin_amdgcn_wmma_f32_16x16x32_bf16(
                    false, fa[mi].v, false, fb[ni].v, (short)0, acc[mi][ni], false, false);

        if (more && MODE == 0) maskStoreB(buf ^ 1);  // fill other buffer (masked)
        wait_async0();
        __syncthreads();
        buf ^= 1;
    }

    // ---- epilogue: D layout -> m_local = vv + 8*h, n_local = lane&15 ----
#pragma unroll
    for (int mi = 0; mi < 4; ++mi) {
        const int gmb = rowbase + wm * 64 + mi * 16 + 8 * h;   // + vv
#pragma unroll
        for (int ni = 0; ni < 2; ++ni) {
            const int gn = colbase + wn * 32 + ni * 16 + m16;
            const size_t ob = (size_t)gmb * VV + gn;
#pragma unroll
            for (int vv = 0; vv < 8; ++vv) {
                float val = acc[mi][ni][vv];
                const size_t o = ob + (size_t)vv * VV;
                if (MODE == 0)      outF[o] = val;
                else if (MODE == 1) outB[o] = f2bf(val / (aux[gmb + vv] * aux[gn]));
                else if (MODE == 2) outB[o] = f2bf(val + aux[o]);
                else if (MODE == 3) outB[o] = f2bf(val + aux[gn]);
                else                outF[o] = val + aux[(gmb + vv) & 31];
            }
        }
    }
}

// ---------------------------------------------------------------------------
// Transpose + convert: in fp32 (R x C) -> out bf16 (C x R)
// ---------------------------------------------------------------------------
__global__ __launch_bounds__(256)
void k_transpose(const float* __restrict__ in, unsigned short* __restrict__ out,
                 int R, int C)
{
    __shared__ float t[64][65];
    const int c0 = blockIdx.x * 64, r0 = blockIdx.y * 64;
#pragma unroll
    for (int i = 0; i < 16; ++i) {
        int idx = threadIdx.x + i * 256;
        int rr = idx >> 6, cc = idx & 63;
        t[rr][cc] = in[(size_t)(r0 + rr) * C + c0 + cc];
    }
    __syncthreads();
#pragma unroll
    for (int i = 0; i < 16; ++i) {
        int idx = threadIdx.x + i * 256;
        int cr = idx >> 6, rc = idx & 63;
        out[(size_t)(c0 + cr) * R + r0 + rc] = f2bf(t[rc][cr]);
    }
}

// count_pos[k] = #{v : x[k,v] > 0}
__global__ void k_count(const float* __restrict__ x, float* __restrict__ cnt)
{
    __shared__ float red[256];
    const size_t k = blockIdx.x;
    const float* row = x + k * VV;
    float c = 0.f;
    for (int i = threadIdx.x; i < VV; i += 256) c += (row[i] > 0.f) ? 1.f : 0.f;
    red[threadIdx.x] = c; __syncthreads();
    for (int s = 128; s > 0; s >>= 1) {
        if (threadIdx.x < s) red[threadIdx.x] += red[threadIdx.x + s];
        __syncthreads();
    }
    if (threadIdx.x == 0) cnt[k] = red[0];
}

// per-variable self-loop: sum over 4 chunks of max(2*diag_c - rowsum_c, 0)
__global__ void k_chunkstats(const unsigned short* __restrict__ xbT,
                             const float* __restrict__ cnt,
                             float* __restrict__ selfloop)
{
    __shared__ float redD[256], redR[256];
    const int b = blockIdx.x;
    const unsigned short* row = xbT + (size_t)b * KTOT;
    float sl = 0.f;
    for (int c = 0; c < 4; ++c) {
        float d = 0.f, rs = 0.f;
        for (int j = threadIdx.x; j < 4096; j += 256) {
            int k = c * 4096 + j;
            float v = bf2f(row[k]);
            if (v > 0.f) d += v;
            rs += v * cnt[k];
        }
        redD[threadIdx.x] = d; redR[threadIdx.x] = rs; __syncthreads();
        for (int s = 128; s > 0; s >>= 1) {
            if (threadIdx.x < s) { redD[threadIdx.x] += redD[threadIdx.x + s];
                                   redR[threadIdx.x] += redR[threadIdx.x + s]; }
            __syncthreads();
        }
        if (threadIdx.x == 0) { float v = 2.f * redD[0] - redR[0]; sl += (v > 0.f) ? v : 0.f; }
        __syncthreads();
    }
    if (threadIdx.x == 0) selfloop[b] = sl;
}

// apply diag fix (in-flight), compute row norms, convert A -> bf16
__global__ void k_diag_norm(const float* __restrict__ A,
                            const float* __restrict__ selfloop,
                            float* __restrict__ norms,
                            unsigned short* __restrict__ Abf)
{
    __shared__ float red[256];
    const int b = blockIdx.x;
    const float* row = A + (size_t)b * VV;
    const float sl = selfloop[b];
    float s2 = 0.f;
    for (int c = threadIdx.x; c < VV; c += 256) {
        float v = (c == b) ? sl : row[c];
        s2 += v * v;
    }
    red[threadIdx.x] = s2; __syncthreads();
    for (int s = 128; s > 0; s >>= 1) {
        if (threadIdx.x < s) red[threadIdx.x] += red[threadIdx.x + s];
        __syncthreads();
    }
    if (threadIdx.x == 0) { float n = sqrtf(red[0]); norms[b] = (n > 1e-12f) ? n : 1e-12f; }
    for (int c = threadIdx.x; c < VV; c += 256) {
        float v = (c == b) ? sl : row[c];
        Abf[(size_t)b * VV + c] = f2bf(v);
    }
}

// time-first projection: z[(s*32+ot), u] = sum_t x[s,t,u] * TW[t,ot]   (bf16 out)
__global__ __launch_bounds__(256)
void k_timeproj(const float* __restrict__ x, const float* __restrict__ TW,
                unsigned short* __restrict__ zb)
{
    __shared__ float tw[TT * OT];
    const int s = blockIdx.x;
    const int u = blockIdx.y * 256 + threadIdx.x;
    for (int i = threadIdx.x; i < TT * OT; i += 256) tw[i] = TW[i];
    __syncthreads();
    float acc[OT];
#pragma unroll
    for (int o = 0; o < OT; ++o) acc[o] = 0.f;
    for (int t = 0; t < TT; ++t) {
        float xv = x[((size_t)s * TT + t) * VV + u];
#pragma unroll
        for (int o = 0; o < OT; ++o) acc[o] += xv * tw[t * OT + o];
    }
#pragma unroll
    for (int o = 0; o < OT; ++o)
        zb[((size_t)s * OT + o) * VV + u] = f2bf(acc[o]);
}

// ---------------------------------------------------------------------------
extern "C" void kernel_launch(void* const* d_in, const int* in_sizes, int n_in,
                              void* d_out, int out_size, void* d_ws, size_t ws_size,
                              hipStream_t stream)
{
    const float* x  = (const float*)d_in[0];
    const float* CW = (const float*)d_in[1];
    const float* CB = (const float*)d_in[2];
    const float* AW = (const float*)d_in[3];
    const float* AB = (const float*)d_in[4];
    const float* TW = (const float*)d_in[5];
    const float* TB = (const float*)d_in[6];
    float* out = (float*)d_out;

    char* w = (char*)d_ws;
    auto alloc = [&](size_t bytes) {
        void* p = (void*)w; w += (bytes + 255) & ~(size_t)255; return p;
    };
    unsigned short* xbT    = (unsigned short*)alloc((size_t)VV * KTOT * 2); // x^T bf16
    unsigned short* Abf    = (unsigned short*)alloc((size_t)VV * VV * 2);
    unsigned short* simb   = (unsigned short*)alloc((size_t)VV * VV * 2);
    unsigned short* simWb  = (unsigned short*)alloc((size_t)VV * VV * 2);
    unsigned short* adj2Tb = (unsigned short*)alloc((size_t)VV * VV * 2);
    unsigned short* CWtb   = (unsigned short*)alloc((size_t)VV * VV * 2);
    unsigned short* AWtb   = (unsigned short*)alloc((size_t)VV * VV * 2);
    unsigned short* zb     = (unsigned short*)alloc((size_t)SS * OT * VV * 2);
    float* Afp   = (float*)alloc((size_t)VV * VV * 4);
    float* cnt   = (float*)alloc((size_t)KTOT * 4);
    float* sloop = (float*)alloc((size_t)VV * 4);
    float* norms = (float*)alloc((size_t)VV * 4);

    // 1) x (16384 x 2048 f32) -> xbT (2048 x 16384 bf16)
    k_transpose<<<dim3(VV / 64, KTOT / 64), 256, 0, stream>>>(x, xbT, KTOT, VV);
    // 2) positive counts per (s,t)
    k_count<<<KTOT, 256, 0, stream>>>(x, cnt);
    // 3) per-chunk diag / rowsum stats -> self-loops
    k_chunkstats<<<VV, 256, 0, stream>>>(xbT, cnt, sloop);
    // 4) adjacency A = X^T * mask(X):  2048 x 2048, K=16384  (WMMA bf16)
    gemm_nt<0, KTOT, KTOT><<<dim3(16, 16), 256, 0, stream>>>(xbT, xbT,
                                                             Afp, nullptr, nullptr);
    // 5) diag fix + row norms + bf16 convert
    k_diag_norm<<<VV, 256, 0, stream>>>(Afp, sloop, norms, Abf);
    // 6) weight transposes (f32 -> bf16)
    k_transpose<<<dim3(32, 32), 256, 0, stream>>>(CW, CWtb, VV, VV);
    k_transpose<<<dim3(32, 32), 256, 0, stream>>>(AW, AWtb, VV, VV);
    // 7) sim = (A A^T) / (n_i n_j)
    gemm_nt<1, VV, VV><<<dim3(16, 16), 256, 0, stream>>>(Abf, Abf,
                                                         nullptr, simb, norms);
    // 8) simW = sim @ CW + CB
    gemm_nt<2, VV, VV><<<dim3(16, 16), 256, 0, stream>>>(simb, CWtb,
                                                         nullptr, simWb, CB);
    // 9) adj2^T[v,u] = sum_c AWt[v,c]*simW[u,c] + AB[u]
    gemm_nt<3, VV, VV><<<dim3(16, 16), 256, 0, stream>>>(AWtb, simWb,
                                                         nullptr, adj2Tb, AB);
    // 10) time-first projection z = einsum('stu,to->sou', x, TW)
    k_timeproj<<<dim3(SS, VV / 256), 256, 0, stream>>>(x, TW, zb);
    // 11) out[(s,ot),v] = sum_u z[(s,ot),u]*adj2[u,v] + tb[ot]   (WMMA bf16)
    gemm_nt<4, VV, VV><<<dim3(16, 32), 256, 0, stream>>>(zb, adj2Tb,
                                                         out, nullptr, TB);
    (void)in_sizes; (void)n_in; (void)out_size; (void)ws_size;
}
